// AttentionWithRoPE_18090402251191
// MI455X (gfx1250) — matrix operand order repaired and checked by reading, not compiled, not run on hardware
//
#include <hip/hip_runtime.h>

// ---------------------------------------------------------------------------
// AttentionWithRoPE for MI455X (gfx1250): bf16 WMMA (16x16x32), f32 accum.
// b=4, n=2048, DIM=512, H=8, Dh=64.
// Attention uses 64-key blocks to amortize softmax VALU over 16 WMMA.
// ---------------------------------------------------------------------------

typedef __attribute__((ext_vector_type(16))) __bf16 v16bf;
typedef __attribute__((ext_vector_type(8)))  float  v8f;

union V16U { v16bf v; uint4 u[2]; };

__device__ __forceinline__ unsigned short f2bf(float f) {
  unsigned int u = __float_as_uint(f);
  u += 0x7FFFu + ((u >> 16) & 1u);     // round-to-nearest-even
  return (unsigned short)(u >> 16);
}

__device__ __forceinline__ v8f wmma_bf16(const V16U& a, const V16U& b, v8f c) {
  return __builtin_amdgcn_wmma_f32_16x16x32_bf16(
      /*neg_a=*/false, a.v, /*neg_b=*/false, b.v,
      /*c_mod=*/(short)0, c, /*reuse_a=*/false, /*reuse_b=*/false);
}

#define NROWS   8192      // b*n
#define DIMK    512
#define QKVCOLS 1536
#define SEQ     2048
#define DH      64

// ------------------------- 1) RMS-norm -> bf16 xn ---------------------------
__global__ void k_rmsnorm(const float* __restrict__ x, const float* __restrict__ gamma,
                          unsigned short* __restrict__ xn) {
  int wave = threadIdx.x >> 5, lane = threadIdx.x & 31;
  int row  = blockIdx.x * 8 + wave;               // 8192 rows, wave per row
  const float* xr = x + (size_t)row * DIMK;
  float vals[16];
  float s = 0.f;
#pragma unroll
  for (int j = 0; j < 16; ++j) { float v = xr[j * 32 + lane]; vals[j] = v; s += v * v; }
#pragma unroll
  for (int m = 1; m < 32; m <<= 1) s += __shfl_xor(s, m, 32);
  float norm = sqrtf(s);
  float sc   = 22.627416997969522f / fmaxf(norm, 1e-8f);   // sqrt(512)/max(norm,eps)
  unsigned short* xo = xn + (size_t)row * DIMK;
#pragma unroll
  for (int j = 0; j < 16; ++j) {
    int c = j * 32 + lane;
    xo[c] = f2bf(vals[j] * sc * gamma[c]);
  }
}

// ------------------ 2) weights -> bf16, transposed [N][K] -------------------
__global__ void k_wconv(const float* __restrict__ wqkv, const float* __restrict__ wout,
                        unsigned short* __restrict__ wTqkv, unsigned short* __restrict__ wTout) {
  int idx = blockIdx.x * blockDim.x + threadIdx.x;   // 512*1536 + 512*512 threads
  if (idx < DIMK * QKVCOLS) {
    int kk = idx / QKVCOLS, n = idx % QKVCOLS;
    wTqkv[(size_t)n * DIMK + kk] = f2bf(wqkv[idx]);
  } else {
    int j  = idx - DIMK * QKVCOLS;
    int kk = j / DIMK, n = j % DIMK;
    wTout[(size_t)n * DIMK + kk] = f2bf(wout[j]);
  }
}

// -------- 3) QKV GEMM (16x64 tile / wave) + RoPE epilogue -------------------
// writes q,k as bf16 [b,h,n,64]; v transposed as bf16 [b,h,64,n]
__global__ void k_qkv_rope(const unsigned short* __restrict__ xn,
                           const unsigned short* __restrict__ wT,
                           unsigned short* __restrict__ qb,
                           unsigned short* __restrict__ kb,
                           unsigned short* __restrict__ vT) {
  int wave = threadIdx.x >> 5, lane = threadIdx.x & 31;
  int gw = blockIdx.x * 8 + wave;                 // 12288 wave-tiles
  int ct = gw % 24, rt = gw / 24;
  int rowbase = rt * 16, colbase = ct * 64;
  int half = lane >> 4, lm = lane & 15;
  int klo = half * 16, kbb = half * 16;

  v8f acc[4];
#pragma unroll
  for (int t = 0; t < 4; ++t)
#pragma unroll
    for (int r = 0; r < 8; ++r) acc[t][r] = 0.f;

  const unsigned short* arow = xn + (size_t)(rowbase + lm) * DIMK;
  for (int kt = 0; kt < 16; ++kt) {
    V16U A;
    A.u[0] = *(const uint4*)(arow + kt * 32 + klo);
    A.u[1] = *(const uint4*)(arow + kt * 32 + klo + 8);
#pragma unroll
    for (int nt = 0; nt < 4; ++nt) {
      const unsigned short* brow = wT + (size_t)(colbase + nt * 16 + lm) * DIMK + kt * 32 + kbb;
      V16U B;
      B.u[0] = *(const uint4*)(brow);
      B.u[1] = *(const uint4*)(brow + 8);
      acc[nt] = wmma_bf16(A, B, acc[nt]);
    }
  }

  int bidx   = rowbase >> 11;          // /2048
  int within = rowbase & 2047;
  int kind   = colbase / 512;          // 0=q, 1=k, 2=v
  int h      = (colbase & 511) >> 6;
  size_t bh  = (size_t)(bidx * 8 + h);

#pragma unroll
  for (int nt = 0; nt < 4; ++nt) {
    int d = nt * 16 + lm;
    float invf = __powf(10000.f, -(float)(d & ~1) * (1.0f / 64.0f));
#pragma unroll
    for (int r = 0; r < 8; ++r) {
      float v = acc[nt][r];
      int m = r + half * 8;
      int npos = within + m;
      if (kind < 2) {                         // RoPE on q,k (wave-uniform branch)
        float pv = __shfl_xor(v, 1, 32);      // partner dim d^1 (adjacent lane)
        float ang = (float)npos * invf;
        float c, s;
        __sincosf(ang, &c, &s);
        v = (d & 1) ? fmaf(v, c, pv * s) : fmaf(v, c, -pv * s);
      }
      unsigned short bv = f2bf(v);
      if (kind == 0)      qb[(bh * SEQ + npos) * DH + d] = bv;
      else if (kind == 1) kb[(bh * SEQ + npos) * DH + d] = bv;
      else                vT[(bh * DH + d) * SEQ + npos] = bv;
    }
  }
}

// --------------------- 4) flash attention (wave = 16 q rows) ----------------
// 64-key blocks: 16 WMMA per softmax update.
__global__ void k_attn(const unsigned short* __restrict__ qb,
                       const unsigned short* __restrict__ kbuf,
                       const unsigned short* __restrict__ vT,
                       unsigned short* __restrict__ ao) {
  __shared__ __align__(16) unsigned short pbuf[8][1024];   // per-wave 16x64 bf16 P tile
  int wave = threadIdx.x >> 5, lane = threadIdx.x & 31;
  int gw = blockIdx.x * 8 + wave;          // 4096 wave-tiles
  int qt = gw & 127;
  int h  = (gw >> 7) & 7;
  int b  = gw >> 10;
  int half = lane >> 4, lm = lane & 15;
  int klo = half * 16, kbb = half * 16;

  size_t bh = (size_t)(b * 8 + h);
  const unsigned short* Q = qb   + bh * SEQ * DH;
  const unsigned short* K = kbuf + bh * SEQ * DH;
  const unsigned short* V = vT   + bh * DH * SEQ;
  int qrow = qt * 16;

  // Q A-tiles (d 0..31 and 32..63), resident in registers
  const unsigned short* qr = Q + (size_t)(qrow + lm) * DH;
  V16U A0, A1;
  A0.u[0] = *(const uint4*)(qr + klo);
  A0.u[1] = *(const uint4*)(qr + klo + 8);
  A1.u[0] = *(const uint4*)(qr + 32 + klo);
  A1.u[1] = *(const uint4*)(qr + 32 + klo + 8);

  float mr[8], lr[8];
  v8f O[4];
#pragma unroll
  for (int r = 0; r < 8; ++r) { mr[r] = -1e30f; lr[r] = 0.f; }
#pragma unroll
  for (int t = 0; t < 4; ++t)
#pragma unroll
    for (int r = 0; r < 8; ++r) O[t][r] = 0.f;

  unsigned short* pw = pbuf[wave];
  const float scale = 0.125f;              // 1/sqrt(64)

  for (int kb0 = 0; kb0 < SEQ; kb0 += 64) {
    // ---- S = Q K^T for 16 q rows x 64 keys: 4 n-tiles x 2 k-tiles ----------
    v8f S[4];
#pragma unroll
    for (int j = 0; j < 4; ++j) {
#pragma unroll
      for (int r = 0; r < 8; ++r) S[j][r] = 0.f;
      const unsigned short* kj = K + (size_t)(kb0 + j * 16 + lm) * DH + kbb;
      V16U Ba, Bb;
      Ba.u[0] = *(const uint4*)(kj);      Ba.u[1] = *(const uint4*)(kj + 8);
      Bb.u[0] = *(const uint4*)(kj + 32); Bb.u[1] = *(const uint4*)(kj + 40);
      S[j] = wmma_bf16(A0, Ba, S[j]);
      S[j] = wmma_bf16(A1, Bb, S[j]);
    }

    // ---- online softmax over 64 keys (row stats per-VGPR, 16-lane halves) --
    float al[8];
#pragma unroll
    for (int r = 0; r < 8; ++r) {
      float s0 = S[0][r] * scale, s1 = S[1][r] * scale;
      float s2 = S[2][r] * scale, s3 = S[3][r] * scale;
      float bm = fmaxf(fmaxf(s0, s1), fmaxf(s2, s3));
#pragma unroll
      for (int msk = 1; msk < 16; msk <<= 1) bm = fmaxf(bm, __shfl_xor(bm, msk, 32));
      float mnew  = fmaxf(mr[r], bm);
      float p0 = __expf(s0 - mnew);
      float p1 = __expf(s1 - mnew);
      float p2 = __expf(s2 - mnew);
      float p3 = __expf(s3 - mnew);
      float alpha = __expf(mr[r] - mnew);
      float ps = (p0 + p1) + (p2 + p3);
#pragma unroll
      for (int msk = 1; msk < 16; msk <<= 1) ps += __shfl_xor(ps, msk, 32);
      lr[r] = lr[r] * alpha + ps;
      mr[r] = mnew;
      al[r] = alpha;
      int m = r + half * 8;                 // C-layout -> row-major P tile in LDS
      pw[m * 64 + lm]      = f2bf(p0);
      pw[m * 64 + lm + 16] = f2bf(p1);
      pw[m * 64 + lm + 32] = f2bf(p2);
      pw[m * 64 + lm + 48] = f2bf(p3);
    }
#pragma unroll
    for (int t = 0; t < 4; ++t)
#pragma unroll
      for (int r = 0; r < 8; ++r) O[t][r] *= al[r];

    asm volatile("s_wait_dscnt 0" ::: "memory");   // same-wave LDS store->load

    // ---- reload P as two A-tiles (keys 0..31 and 32..63) -------------------
    V16U P0, P1;
    const unsigned short* pr = pw + lm * 64 + klo;
    P0.u[0] = *(const uint4*)(pr);
    P0.u[1] = *(const uint4*)(pr + 8);
    P1.u[0] = *(const uint4*)(pr + 32);
    P1.u[1] = *(const uint4*)(pr + 40);

    // ---- O += P V : 4 d-tiles x 2 key k-tiles ------------------------------
#pragma unroll
    for (int t = 0; t < 4; ++t) {
      const unsigned short* vr = V + (size_t)(t * 16 + lm) * SEQ + kb0 + kbb;
      V16U Bv0, Bv1;
      Bv0.u[0] = *(const uint4*)(vr);      Bv0.u[1] = *(const uint4*)(vr + 8);
      Bv1.u[0] = *(const uint4*)(vr + 32); Bv1.u[1] = *(const uint4*)(vr + 40);
      O[t] = wmma_bf16(P0, Bv0, O[t]);
      O[t] = wmma_bf16(P1, Bv1, O[t]);
    }
  }

  float il[8];
#pragma unroll
  for (int r = 0; r < 8; ++r) il[r] = 1.f / lr[r];
#pragma unroll
  for (int t = 0; t < 4; ++t) {
    int d = t * 16 + lm;
#pragma unroll
    for (int r = 0; r < 8; ++r) {
      int m = r + half * 8;
      int npos = qrow + m;
      ao[((size_t)(b * SEQ + npos)) * 512 + h * 64 + d] = f2bf(O[t][r] * il[r]);
    }
  }
}

// ------------------------- 5) out projection (f32 out) ----------------------
__global__ void k_outproj(const unsigned short* __restrict__ ain,
                          const unsigned short* __restrict__ wT,
                          float* __restrict__ out) {
  int wave = threadIdx.x >> 5, lane = threadIdx.x & 31;
  int gw = blockIdx.x * 8 + wave;          // 4096 wave-tiles
  int ct = gw & 7, rt = gw >> 3;
  int rowbase = rt * 16, colbase = ct * 64;
  int half = lane >> 4, lm = lane & 15;
  int klo = half * 16, kbb = half * 16;

  v8f acc[4];
#pragma unroll
  for (int t = 0; t < 4; ++t)
#pragma unroll
    for (int r = 0; r < 8; ++r) acc[t][r] = 0.f;

  const unsigned short* arow = ain + (size_t)(rowbase + lm) * DIMK;
  for (int kt = 0; kt < 16; ++kt) {
    V16U A;
    A.u[0] = *(const uint4*)(arow + kt * 32 + klo);
    A.u[1] = *(const uint4*)(arow + kt * 32 + klo + 8);
#pragma unroll
    for (int nt = 0; nt < 4; ++nt) {
      const unsigned short* brow = wT + (size_t)(colbase + nt * 16 + lm) * DIMK + kt * 32 + kbb;
      V16U B;
      B.u[0] = *(const uint4*)(brow);
      B.u[1] = *(const uint4*)(brow + 8);
      acc[nt] = wmma_bf16(A, B, acc[nt]);
    }
  }
#pragma unroll
  for (int nt = 0; nt < 4; ++nt) {
    int col = colbase + nt * 16 + lm;
#pragma unroll
    for (int r = 0; r < 8; ++r) {
      int row = rowbase + r + half * 8;
      out[(size_t)row * 512 + col] = acc[nt][r];
    }
  }
}

// ---------------------------------------------------------------------------
extern "C" void kernel_launch(void* const* d_in, const int* in_sizes, int n_in,
                              void* d_out, int out_size, void* d_ws, size_t ws_size,
                              hipStream_t stream) {
  const float* x     = (const float*)d_in[0];   // [4,2048,512]
  const float* gamma = (const float*)d_in[1];   // [512]
  const float* wqkv  = (const float*)d_in[2];   // [512,1536]
  const float* wout  = (const float*)d_in[3];   // [512,512]
  float* out = (float*)d_out;                   // [4,2048,512]

  char* ws = (char*)d_ws;
  size_t off = 0;
  auto carve = [&](size_t bytes) { char* p = ws + off; off += (bytes + 255) & ~(size_t)255; return p; };
  unsigned short* xn    = (unsigned short*)carve((size_t)NROWS * DIMK * 2);     // 8 MB
  unsigned short* wTqkv = (unsigned short*)carve((size_t)QKVCOLS * DIMK * 2);   // 1.5 MB
  unsigned short* wTout = (unsigned short*)carve((size_t)DIMK * DIMK * 2);      // 0.5 MB
  unsigned short* qbuf  = (unsigned short*)carve((size_t)NROWS * DIMK * 2);     // 8 MB
  unsigned short* kbuf  = (unsigned short*)carve((size_t)NROWS * DIMK * 2);     // 8 MB
  unsigned short* vTb   = (unsigned short*)carve((size_t)NROWS * DIMK * 2);     // 8 MB (transposed)
  unsigned short* attn  = (unsigned short*)carve((size_t)NROWS * DIMK * 2);     // 8 MB
  (void)ws_size; (void)in_sizes; (void)n_in; (void)out_size;

  k_rmsnorm <<<NROWS / 8, 256, 0, stream>>>(x, gamma, xn);
  k_wconv   <<<(DIMK * QKVCOLS + DIMK * DIMK) / 256, 256, 0, stream>>>(wqkv, wout, wTqkv, wTout);
  k_qkv_rope<<<(NROWS / 16) * (QKVCOLS / 64) / 8, 256, 0, stream>>>(xn, wTqkv, qbuf, kbuf, vTb);
  k_attn    <<<(4 * 8 * (SEQ / 16)) / 8, 256, 0, stream>>>(qbuf, kbuf, vTb, attn);
  k_outproj <<<(NROWS / 16) * (DIMK / 64) / 8, 256, 0, stream>>>(attn, wTout, out);
}
